// DirectionalGAT_24421184045779
// MI455X (gfx1250) — matrix operand order repaired and checked by reading, not compiled, not run on hardware
//
#include <hip/hip_runtime.h>

typedef __attribute__((ext_vector_type(16))) _Float16 v16h;
typedef __attribute__((ext_vector_type(8)))  _Float16 v8h;
typedef __attribute__((ext_vector_type(8)))  float    v8f;

#define WMMA_F16(a,b,c) __builtin_amdgcn_wmma_f32_16x16x32_f16(false,(a),false,(b),(short)0,(c),false,false)

// ---------------- helpers ----------------
__device__ __forceinline__ unsigned fkey(float x){
  unsigned u = __float_as_uint(x);
  return (u & 0x80000000u) ? ~u : (u | 0x80000000u);
}
__device__ __forceinline__ float funkey(unsigned u){
  return (u & 0x80000000u) ? __uint_as_float(u & 0x7FFFFFFFu) : __uint_as_float(~u);
}
__device__ __forceinline__ void atomAddF(float* p, float v){ unsafeAtomicAdd(p, v); }

// Build a 16-half A fragment from two 8-float global segments (nullptr -> zeros)
__device__ __forceinline__ v16h make_a(const float* p0, const float* p1){
  float f[16];
  if (p0){
    float4 a = ((const float4*)p0)[0], b = ((const float4*)p0)[1];
    f[0]=a.x; f[1]=a.y; f[2]=a.z; f[3]=a.w; f[4]=b.x; f[5]=b.y; f[6]=b.z; f[7]=b.w;
  } else {
    #pragma unroll
    for (int i=0;i<8;i++) f[i]=0.f;
  }
  if (p1){
    float4 a = ((const float4*)p1)[0], b = ((const float4*)p1)[1];
    f[8]=a.x; f[9]=a.y; f[10]=a.z; f[11]=a.w; f[12]=b.x; f[13]=b.y; f[14]=b.z; f[15]=b.w;
  } else {
    #pragma unroll
    for (int i=0;i<8;i++) f[8+i]=0.f;
  }
  v16h r;
  #pragma unroll
  for (int i=0;i<16;i++) r[i] = (_Float16)f[i];
  return r;
}

// A fragment from 16-half-per-lane LDS row (stride in halves)
__device__ __forceinline__ v16h lds_frag(const _Float16* rowbase, int kc, int hi){
  v8h lo = *(const v8h*)(rowbase + kc*32 + 8*hi);
  v8h hv = *(const v8h*)(rowbase + kc*32 + 16 + 8*hi);
  v16h r;
  #pragma unroll
  for (int i=0;i<8;i++){ r[i] = lo[i]; r[8+i] = hv[i]; }
  return r;
}

// map concatenated edge-feature column k -> global pointer (8-float segment)
__device__ __forceinline__ const float* edge_feat_ptr(int k, const float* x, const float* xs,
                                                      const float* ef, int s, int d, int e){
  if (k < 96)  return x  + (size_t)s*96 + k;
  if (k < 192) return x  + (size_t)d*96 + (k-96);
  if (k < 240) return xs + (size_t)s*48 + (k-192);
  if (k < 288) return xs + (size_t)d*48 + (k-240);
  if (k < 312) return ef + (size_t)e*24 + (k-288);
  return nullptr;
}

__device__ __forceinline__ const float* gate_ptr(int k, const float* x, const float* mf,
                                                 const float* mr, int node){
  if (k < 96)  return x  + (size_t)node*96 + k;
  if (k < 192) return mf + (size_t)node*96 + (k-96);
  return mr + (size_t)node*96 + (k-192);
}

// ---------------- weight prep ----------------
__global__ void cvt_f16_kernel(const float* __restrict__ src, _Float16* __restrict__ dst, int n){
  int i = blockIdx.x*blockDim.x + threadIdx.x;
  if (i < n) dst[i] = (_Float16)src[i];
}

// edge W1 (192x312) -> f16 (192x320), optional K-permutation for the reverse MLP
__global__ void edge_w1_kernel(const float* __restrict__ w, _Float16* __restrict__ dst, int permute){
  int i = blockIdx.x*blockDim.x + threadIdx.x;
  if (i >= 192*320) return;
  int j = i/320, k = i%320;
  float v = 0.f;
  if (k < 312){
    int ks = k;
    if (permute){
      if (k < 96)       ks = k + 96;
      else if (k < 192) ks = k - 96;
      else if (k < 240) ks = k + 48;
      else if (k < 288) ks = k - 48;
    }
    v = w[j*312 + ks];
  }
  dst[i] = (_Float16)v;
}

// ---------------- edge scoring: 1 wave = 16 edges, WMMA 312->192, fold 192->1 ----------------
__global__ __launch_bounds__(256) void edge_score_kernel(
    const float* __restrict__ x, const float* __restrict__ xs, const float* __restrict__ ef,
    const int* __restrict__ src, const int* __restrict__ dst,
    const _Float16* __restrict__ W1F, const _Float16* __restrict__ W1R,
    const float* __restrict__ b1F, const float* __restrict__ w2F, const float* __restrict__ b2F,
    const float* __restrict__ b1R, const float* __restrict__ w2R, const float* __restrict__ b2R,
    float* __restrict__ sF, float* __restrict__ revw, unsigned* __restrict__ nodeMax,
    int E, int ntiles)
{
  int wave = threadIdx.x >> 5, lane = threadIdx.x & 31;
  int tile = blockIdx.x * (blockDim.x >> 5) + wave;
  if (tile >= ntiles) return;
  int e0 = tile * 16;
  int row = lane & 15, hi = lane >> 4, col = row;
  int er = e0 + row; if (er > E-1) er = E-1;
  int s = src[er], d = dst[er];

  v16h A[10];
  #pragma unroll
  for (int c=0;c<10;c++){
    int k1 = c*32 + 8*hi, k2 = c*32 + 16 + 8*hi;
    A[c] = make_a(edge_feat_ptr(k1,x,xs,ef,s,d,er), edge_feat_ptr(k2,x,xs,ef,s,d,er));
  }
  const float rs = rsqrtf(96.0f);

  for (int dir=0; dir<2; dir++){
    const _Float16* W1 = dir ? W1R : W1F;
    const float* b1 = dir ? b1R : b1F;
    const float* w2 = dir ? w2R : w2F;
    float b2 = dir ? b2R[0] : b2F[0];
    float p[8];
    #pragma unroll
    for (int v=0;v<8;v++) p[v]=0.f;

    for (int nt=0; nt<12; nt++){
      v8f acc = {};
      const _Float16* wb = W1 + (size_t)(nt*16+col)*320 + hi*8;
      #pragma unroll
      for (int kc=0;kc<10;kc++){
        v16h b = __builtin_shufflevector(*(const v8h*)(wb + kc*32), *(const v8h*)(wb + kc*32 + 16), 0,1,2,3,4,5,6,7,8,9,10,11,12,13,14,15);
        acc = WMMA_F16(A[kc], b, acc);
      }
      float bb = b1[nt*16+col], ww = w2[nt*16+col];
      #pragma unroll
      for (int v=0;v<8;v++){
        float h = acc[v] + bb; h = h>0.f ? h : 0.f;
        p[v] += h * ww;
      }
    }
    #pragma unroll
    for (int v=0;v<8;v++){
      #pragma unroll
      for (int off=1; off<16; off<<=1) p[v] += __shfl_xor(p[v], off, 32);
    }
    if (col == 0){
      #pragma unroll
      for (int v=0;v<8;v++){
        int e = e0 + v + 8*hi;
        if (e < E){
          float raw = p[v] + b2;
          if (dir == 0){
            float sc = (raw > 0.f ? raw : 0.01f*raw) * rs;   // leaky_relu / sqrt(H)
            sF[e] = sc;
            atomicMax(nodeMax + dst[e], fkey(sc));
          } else {
            revw[e] = 1.f / (1.f + __expf(-raw));            // sigmoid gate
          }
        }
      }
    }
  }
}

// ---------------- softmax exp + sum ----------------
__global__ void exp_kernel(float* __restrict__ sF, const unsigned* __restrict__ nodeMax,
                           float* __restrict__ nodeSum, const int* __restrict__ dst, int E){
  int e = blockIdx.x*blockDim.x + threadIdx.x;
  if (e >= E) return;
  int d = dst[e];
  float ex = __expf(sF[e] - funkey(nodeMax[d]));
  sF[e] = ex;
  atomAddF(nodeSum + d, ex);
}

// ---------------- normalize + both scatter aggregations: 1 wave = 1 edge ----------------
__global__ __launch_bounds__(256) void agg_kernel(
    const float* __restrict__ x, const int* __restrict__ src, const int* __restrict__ dst,
    float* __restrict__ fwdw, const float* __restrict__ nodeSum, const float* __restrict__ revw,
    float* __restrict__ msgF, float* __restrict__ msgR, int E)
{
  int e = blockIdx.x*(blockDim.x>>5) + (threadIdx.x>>5);
  int lane = threadIdx.x & 31;
  if (e >= E) return;
  int s = src[e], d = dst[e];
  float wF = fwdw[e] / (nodeSum[d] + 1e-9f);
  if (lane == 0) fwdw[e] = wF;     // finalize fwd_w output (all lanes already read old value)
  float wR = revw[e];
  #pragma unroll
  for (int c = lane; c < 96; c += 32){
    atomAddF(msgF + (size_t)d*96 + c, x[(size_t)s*96 + c] * wF);
    atomAddF(msgR + (size_t)s*96 + c, x[(size_t)d*96 + c] * wR);
  }
}

// ---------------- node GRU MLPs: 1 wave = 16 nodes ----------------
__device__ __forceinline__ void node_layer1(const v16h* A, const _Float16* __restrict__ W1,
                                            const float* __restrict__ b1, _Float16* hbuf, int lane){
  int col = lane & 15, hi = lane >> 4;
  for (int nt=0; nt<18; nt++){
    v8f acc = {};
    const _Float16* wb = W1 + (size_t)(nt*16+col)*288 + hi*8;
    #pragma unroll
    for (int kc=0;kc<9;kc++){
      v16h b = __builtin_shufflevector(*(const v8h*)(wb + kc*32), *(const v8h*)(wb + kc*32 + 16), 0,1,2,3,4,5,6,7,8,9,10,11,12,13,14,15);
      acc = WMMA_F16(A[kc], b, acc);
    }
    float bb = b1[nt*16+col];
    #pragma unroll
    for (int v=0;v<8;v++){
      float h = acc[v] + bb; h = h>0.f ? h : 0.f;
      hbuf[(v + 8*hi)*288 + nt*16 + col] = (_Float16)h;
    }
  }
}

__global__ __launch_bounds__(32) void node_kernel(
    const float* __restrict__ x, const float* __restrict__ msgF, const float* __restrict__ msgR,
    const _Float16* __restrict__ W1r, const _Float16* __restrict__ W2r,
    const float* __restrict__ b1r, const float* __restrict__ b2r,
    const _Float16* __restrict__ W1z, const _Float16* __restrict__ W2z,
    const float* __restrict__ b1z, const float* __restrict__ b2z,
    const _Float16* __restrict__ W1c, const _Float16* __restrict__ W2c,
    const float* __restrict__ b1c, const float* __restrict__ b2c,
    float* __restrict__ out_upd, float* __restrict__ out_z, float* __restrict__ out_r, int N)
{
  __shared__ __align__(32) _Float16 hbuf[16*288];
  __shared__ __align__(32) _Float16 rxbuf[16*96];
  int lane = threadIdx.x & 31;
  int row = lane & 15, hi = lane >> 4, col = row;
  int tb = blockIdx.x * 16;
  int node = tb + row; if (node > N-1) node = N-1;

  // gate input A: [x | msgF | msgR] = 288 features, 9 K-chunks
  v16h AG[9];
  #pragma unroll
  for (int c=0;c<9;c++){
    int k1 = c*32 + 8*hi, k2 = c*32 + 16 + 8*hi;
    AG[c] = make_a(gate_ptr(k1,x,msgF,msgR,node), gate_ptr(k2,x,msgF,msgR,node));
  }

  float zsave[6][8];

  // ---- reset gate r ----
  node_layer1(AG, W1r, b1r, hbuf, lane);
  __syncthreads();
  for (int ot=0; ot<6; ot++){
    v8f acc = {};
    const _Float16* wb = W2r + (size_t)(ot*16+col)*288 + hi*8;
    #pragma unroll
    for (int kc=0;kc<9;kc++){
      v16h a = lds_frag(hbuf + row*288, kc, hi);
      v16h b = __builtin_shufflevector(*(const v8h*)(wb + kc*32), *(const v8h*)(wb + kc*32 + 16), 0,1,2,3,4,5,6,7,8,9,10,11,12,13,14,15);
      acc = WMMA_F16(a, b, acc);
    }
    float bb = b2r[ot*16+col];
    #pragma unroll
    for (int v=0;v<8;v++){
      int m = v + 8*hi, j = ot*16 + col;
      float rv = 1.f/(1.f + __expf(-(acc[v] + bb)));
      int nm = tb + m; int nmc = nm > N-1 ? N-1 : nm;
      if (nm < N) out_r[(size_t)nm*96 + j] = rv;
      float xv = x[(size_t)nmc*96 + j];
      rxbuf[m*96 + j] = (_Float16)(rv * xv);
    }
  }
  __syncthreads();

  // ---- update gate z ----
  node_layer1(AG, W1z, b1z, hbuf, lane);
  __syncthreads();
  for (int ot=0; ot<6; ot++){
    v8f acc = {};
    const _Float16* wb = W2z + (size_t)(ot*16+col)*288 + hi*8;
    #pragma unroll
    for (int kc=0;kc<9;kc++){
      v16h a = lds_frag(hbuf + row*288, kc, hi);
      v16h b = __builtin_shufflevector(*(const v8h*)(wb + kc*32), *(const v8h*)(wb + kc*32 + 16), 0,1,2,3,4,5,6,7,8,9,10,11,12,13,14,15);
      acc = WMMA_F16(a, b, acc);
    }
    float bb = b2z[ot*16+col];
    #pragma unroll
    for (int v=0;v<8;v++){
      int m = v + 8*hi, j = ot*16 + col;
      float zv = 1.f/(1.f + __expf(-(acc[v] + bb)));
      zsave[ot][v] = zv;
      if (tb + m < N) out_z[(size_t)(tb+m)*96 + j] = zv;
    }
  }
  __syncthreads();

  // ---- candidate: A = [r*x | msgF | msgR]; chunks 3..8 reuse AG ----
  v16h AC[9];
  #pragma unroll
  for (int c=0;c<3;c++) AC[c] = lds_frag(rxbuf + row*96, c, hi);
  #pragma unroll
  for (int c=3;c<9;c++) AC[c] = AG[c];
  node_layer1(AC, W1c, b1c, hbuf, lane);
  __syncthreads();
  for (int ot=0; ot<6; ot++){
    v8f acc = {};
    const _Float16* wb = W2c + (size_t)(ot*16+col)*288 + hi*8;
    #pragma unroll
    for (int kc=0;kc<9;kc++){
      v16h a = lds_frag(hbuf + row*288, kc, hi);
      v16h b = __builtin_shufflevector(*(const v8h*)(wb + kc*32), *(const v8h*)(wb + kc*32 + 16), 0,1,2,3,4,5,6,7,8,9,10,11,12,13,14,15);
      acc = WMMA_F16(a, b, acc);
    }
    float bb = b2c[ot*16+col];
    #pragma unroll
    for (int v=0;v<8;v++){
      int m = v + 8*hi, j = ot*16 + col;
      if (tb + m < N){
        float cv = tanhf(acc[v] + bb);
        float zv = zsave[ot][v];
        float xv = x[(size_t)(tb+m)*96 + j];
        out_upd[(size_t)(tb+m)*96 + j] = (1.f - zv)*xv + zv*cv;
      }
    }
  }
}

// ---------------- host ----------------
extern "C" void kernel_launch(void* const* d_in, const int* in_sizes, int n_in,
                              void* d_out, int out_size, void* d_ws, size_t ws_size,
                              hipStream_t stream)
{
  const float* x   = (const float*)d_in[0];
  const float* xs  = (const float*)d_in[1];
  const float* ef  = (const float*)d_in[2];
  const int*  eidx = (const int*)d_in[3];
  const float* fwd_W1 = (const float*)d_in[4];  const float* fwd_b1 = (const float*)d_in[5];
  const float* fwd_W2 = (const float*)d_in[6];  const float* fwd_b2 = (const float*)d_in[7];
  const float* rev_W1 = (const float*)d_in[8];  const float* rev_b1 = (const float*)d_in[9];
  const float* rev_W2 = (const float*)d_in[10]; const float* rev_b2 = (const float*)d_in[11];
  const float* rst_W1 = (const float*)d_in[12]; const float* rst_b1 = (const float*)d_in[13];
  const float* rst_W2 = (const float*)d_in[14]; const float* rst_b2 = (const float*)d_in[15];
  const float* upd_W1 = (const float*)d_in[16]; const float* upd_b1 = (const float*)d_in[17];
  const float* upd_W2 = (const float*)d_in[18]; const float* upd_b2 = (const float*)d_in[19];
  const float* cnd_W1 = (const float*)d_in[20]; const float* cnd_b1 = (const float*)d_in[21];
  const float* cnd_W2 = (const float*)d_in[22]; const float* cnd_b2 = (const float*)d_in[23];

  const int N = in_sizes[0] / 96;
  const int E = in_sizes[3] / 2;
  const int* src = eidx;
  const int* dst = eidx + E;

  // workspace carve-up (256B aligned)
  char* w = (char*)d_ws;
  auto carve = [&](size_t bytes)->char*{ char* p = w; w += (bytes + 255) & ~(size_t)255; return p; };
  _Float16* W1F  = (_Float16*)carve(192*320*sizeof(_Float16));
  _Float16* W1R  = (_Float16*)carve(192*320*sizeof(_Float16));
  _Float16* nW1r = (_Float16*)carve(288*288*sizeof(_Float16));
  _Float16* nW1z = (_Float16*)carve(288*288*sizeof(_Float16));
  _Float16* nW1c = (_Float16*)carve(288*288*sizeof(_Float16));
  _Float16* nW2r = (_Float16*)carve(96*288*sizeof(_Float16));
  _Float16* nW2z = (_Float16*)carve(96*288*sizeof(_Float16));
  _Float16* nW2c = (_Float16*)carve(96*288*sizeof(_Float16));
  unsigned* nodeMax = (unsigned*)carve((size_t)N*sizeof(unsigned));
  float*    nodeSum = (float*)carve((size_t)N*sizeof(float));
  float*    msgF    = (float*)carve((size_t)N*96*sizeof(float));
  float*    msgR    = (float*)carve((size_t)N*96*sizeof(float));

  // output layout: update[N,96] | fwd_w[E] | rev_w[E] | z[N,96] | r[N,96]
  float* out_upd = (float*)d_out;
  float* out_fw  = out_upd + (size_t)N*96;
  float* out_rw  = out_fw + E;
  float* out_z   = out_rw + E;
  float* out_r   = out_z + (size_t)N*96;

  // weight prep (f16 conversions; rev W1 K-permuted so one A matrix serves both MLPs)
  edge_w1_kernel<<<(192*320+255)/256, 256, 0, stream>>>(fwd_W1, W1F, 0);
  edge_w1_kernel<<<(192*320+255)/256, 256, 0, stream>>>(rev_W1, W1R, 1);
  cvt_f16_kernel<<<(288*288+255)/256, 256, 0, stream>>>(rst_W1, nW1r, 288*288);
  cvt_f16_kernel<<<(288*288+255)/256, 256, 0, stream>>>(upd_W1, nW1z, 288*288);
  cvt_f16_kernel<<<(288*288+255)/256, 256, 0, stream>>>(cnd_W1, nW1c, 288*288);
  cvt_f16_kernel<<<(96*288+255)/256, 256, 0, stream>>>(rst_W2, nW2r, 96*288);
  cvt_f16_kernel<<<(96*288+255)/256, 256, 0, stream>>>(upd_W2, nW2z, 96*288);
  cvt_f16_kernel<<<(96*288+255)/256, 256, 0, stream>>>(cnd_W2, nW2c, 96*288);

  hipMemsetAsync(nodeMax, 0, (size_t)N*sizeof(unsigned), stream);
  hipMemsetAsync(nodeSum, 0, (size_t)N*sizeof(float), stream);
  hipMemsetAsync(msgF, 0, (size_t)N*96*sizeof(float), stream);
  hipMemsetAsync(msgR, 0, (size_t)N*96*sizeof(float), stream);

  const int etiles = (E + 15) / 16;
  edge_score_kernel<<<(etiles + 7)/8, 256, 0, stream>>>(
      x, xs, ef, src, dst, W1F, W1R,
      fwd_b1, fwd_W2, fwd_b2, rev_b1, rev_W2, rev_b2,
      out_fw, out_rw, nodeMax, E, etiles);

  exp_kernel<<<(E + 255)/256, 256, 0, stream>>>(out_fw, nodeMax, nodeSum, dst, E);

  agg_kernel<<<(E + 7)/8, 256, 0, stream>>>(x, src, dst, out_fw, nodeSum, out_rw, msgF, msgR, E);

  const int ntilesN = (N + 15) / 16;
  node_kernel<<<ntilesN, 32, 0, stream>>>(
      x, msgF, msgR,
      nW1r, nW2r, rst_b1, rst_b2,
      nW1z, nW2z, upd_b1, upd_b2,
      nW1c, nW2c, cnd_b1, cnd_b2,
      out_upd, out_z, out_r, N);
}
